// WubuBlock_49916109914617
// MI455X (gfx1250) — matrix operand order repaired and checked by reading, not compiled, not run on hardware
//
#include <hip/hip_runtime.h>
#include <hip/hip_bf16.h>

typedef __bf16 bf16_t;
typedef __attribute__((ext_vector_type(16))) __bf16 v16bf;
typedef __attribute__((ext_vector_type(8)))  float  v8f;

#define EPSF 1e-7f

// ---------- small device helpers ----------
__device__ __forceinline__ bf16_t f2bf(float f) {
  unsigned u = __float_as_uint(f);
  unsigned r = (u + 0x7FFFu + ((u >> 16) & 1u)) >> 16;
  unsigned short s = (unsigned short)r;
  return __builtin_bit_cast(bf16_t, s);
}
__device__ __forceinline__ float fast_tanh(float x) {
#if __has_builtin(__builtin_amdgcn_tanhf)
  return __builtin_amdgcn_tanhf(x);            // v_tanh_f32 on gfx1250
#else
  return tanhf(x);
#endif
}
__device__ __forceinline__ float atanh_f(float x) {           // x in [0, 1-eps]
  return 0.5f * __logf((1.f + x) / (1.f - x));
}
__device__ __forceinline__ float softplus_f(float x) {
  return (x > 20.f) ? x : __logf(1.f + __expf(x));
}
__device__ __forceinline__ float gelu_f(float x) {            // tanh approximation (jax default)
  const float k0 = 0.7978845608028654f;
  return 0.5f * x * (1.f + fast_tanh(k0 * (x + 0.044715f * x * x * x)));
}

template <int NT>
__device__ __forceinline__ float block_sum(float v, float* sm) {
#pragma unroll
  for (int off = 16; off > 0; off >>= 1) v += __shfl_xor(v, off, 32);
  if constexpr (NT > 32) {
    int lane = threadIdx.x & 31, wid = threadIdx.x >> 5;
    if (lane == 0) sm[wid] = v;
    __syncthreads();
    if (threadIdx.x == 0) {
      float r = sm[0];
#pragma unroll
      for (int i = 1; i < NT / 32; ++i) r += sm[i];
      sm[0] = r;
    }
    __syncthreads();
    v = sm[0];
    __syncthreads();
  }
  return v;
}
template <int NT>
__device__ __forceinline__ float block_max(float v, float* sm) {
#pragma unroll
  for (int off = 16; off > 0; off >>= 1) v = fmaxf(v, __shfl_xor(v, off, 32));
  if constexpr (NT > 32) {
    int lane = threadIdx.x & 31, wid = threadIdx.x >> 5;
    if (lane == 0) sm[wid] = v;
    __syncthreads();
    if (threadIdx.x == 0) {
      float r = sm[0];
#pragma unroll
      for (int i = 1; i < NT / 32; ++i) r = fmaxf(r, sm[i]);
      sm[0] = r;
    }
    __syncthreads();
    v = sm[0];
    __syncthreads();
  }
  return v;
}

// ---------- WMMA fragment loaders (16x16x32 bf16, this code's contiguous k = 16h + s order) ----------
// A: 16x32 (MxK). lanes 0-15: M=lane, h[0..15]=K0..15
//                 lanes16-31: M=lane-16, h[0..15]=K16..31
__device__ __forceinline__ v16bf load_a_frag(const bf16_t* __restrict__ A, int lda,
                                             int m0, int k0, int lane) {
  int m = lane & 15;
  int kb = (lane & 16) ? 16 : 0;
  const bf16_t* p = A + (size_t)(m0 + m) * lda + k0 + kb;
  v16bf a;
#pragma unroll
  for (int j = 0; j < 16; ++j) a[j] = p[j];
  return a;
}
// B fragment from B^T storage (Bt[n][k] row-major, leading dim ldbt) -> contiguous b128 loads
__device__ __forceinline__ v16bf load_bT_frag(const bf16_t* __restrict__ Bt, int ldbt,
                                              int k0, int n0, int lane) {
  int n = lane & 15;
  int kb = (lane & 16) ? 16 : 0;
  const bf16_t* p = Bt + (size_t)(n0 + n) * ldbt + k0 + kb;
  v16bf b;
#pragma unroll
  for (int j = 0; j < 16; ++j) b[j] = p[j];
  return b;
}
__device__ __forceinline__ v8f wmma_bf16(v16bf a, v16bf b, v8f c) {
  return __builtin_amdgcn_wmma_f32_16x16x32_bf16(false, a, false, b, (short)0, c,
                                                 false, false);
}

// ---------- kernels ----------
// convert f32 [K,N] -> bf16 transposed [N,K] via LDS tile (grid: (N/32, K/32), 256 thr)
__global__ void cvtT_k(const float* __restrict__ in, bf16_t* __restrict__ outT,
                       int K, int N) {
  __shared__ float tile[32][33];
  int kb = blockIdx.y * 32, nb = blockIdx.x * 32;
  int tx = threadIdx.x & 31, ty = threadIdx.x >> 5;   // 32 x 8
#pragma unroll
  for (int i = ty; i < 32; i += 8)
    tile[i][tx] = in[(size_t)(kb + i) * N + nb + tx];
  __syncthreads();
#pragma unroll
  for (int i = ty; i < 32; i += 8)
    outT[(size_t)(nb + i) * K + kb + tx] = f2bf(tile[tx][i]);
}

// per-token: logmap0 -> layernorm(ln1) -> expmap0(+project) -> logmap0  => xt (bf16)
__global__ void prep_k(const float* __restrict__ x, const float* __restrict__ c_global,
                       const float* __restrict__ ln_s, const float* __restrict__ ln_b,
                       bf16_t* __restrict__ xt_bf) {
  int row = blockIdx.x;          // 0..1023
  int t = threadIdx.x;           // 256 threads, 2 elems each
  __shared__ float sm[8];
  const float* xr = x + (size_t)row * 512;
  float v0 = xr[t], v1 = xr[t + 256];
  float c = c_global[0];
  float sc = fmaxf(sqrtf(c), EPSF);
  // logmap0
  float yn = sqrtf(block_sum<256>(v0 * v0 + v1 * v1, sm));
  float mag = atanh_f(fminf(yn, 1.f - EPSF)) / sc;
  float s1 = (yn < EPSF) ? 0.f : mag / fmaxf(yn, EPSF);
  float u0 = v0 * s1, u1 = v1 * s1;
  // layernorm
  float mu = block_sum<256>(u0 + u1, sm) * (1.f / 512.f);
  float d0 = u0 - mu, d1 = u1 - mu;
  float var = block_sum<256>(d0 * d0 + d1 * d1, sm) * (1.f / 512.f);
  float rs = rsqrtf(var + 1e-6f);
  float w0 = d0 * rs * ln_s[t] + ln_b[t];
  float w1 = d1 * rs * ln_s[t + 256] + ln_b[t + 256];
  // expmap0 + project
  float vn = sqrtf(block_sum<256>(w0 * w0 + w1 * w1, sm));
  float mg = fast_tanh(sc * fmaxf(vn, EPSF)) / sc;
  float s2 = (vn < EPSF) ? 0.f : mg / fmaxf(vn, EPSF);
  float e0 = w0 * s2, e1 = w1 * s2;
  float en = (vn < EPSF) ? 0.f : mg;
  float ps = (en >= 1.f) ? (1.f - EPSF) / fmaxf(en, EPSF) : 1.f;
  e0 *= ps; e1 *= ps;
  float enp = (en >= 1.f) ? (1.f - EPSF) : en;
  // logmap0 back to tangent
  float mg2 = atanh_f(fminf(enp, 1.f - EPSF)) / sc;
  float s3 = (enp < EPSF) ? 0.f : mg2 / fmaxf(enp, EPSF);
  xt_bf[(size_t)row * 512 + t] = f2bf(e0 * s3);
  xt_bf[(size_t)row * 512 + t + 256] = f2bf(e1 * s3);
}

// templated bf16 WMMA GEMM: C[M,N] = act(A[M,K] @ Bt^T + bias)
// Bt is [N,K] (transposed weights). 256 thr = 8 waves; wave owns a 16x32 strip
// (two 16x16 N-tiles sharing one A fragment). Block tile 32x128. Grid (N/128, M/32).
template <int K, int ACT, bool F32OUT, bool BFOUT>
__global__ void __launch_bounds__(256) gemm_tpl_k(const bf16_t* __restrict__ A,
                                                  const bf16_t* __restrict__ Bt,
                                                  const float* __restrict__ bias,
                                                  float* __restrict__ Cf,
                                                  bf16_t* __restrict__ Cb,
                                                  int M, int N) {
  int lane = threadIdx.x & 31;
  int wid = threadIdx.x >> 5;
  int m0 = blockIdx.y * 32 + (wid >> 2) * 16;
  int n0 = blockIdx.x * 128 + (wid & 3) * 32;
  v8f acc0 = {0.f, 0.f, 0.f, 0.f, 0.f, 0.f, 0.f, 0.f};
  v8f acc1 = {0.f, 0.f, 0.f, 0.f, 0.f, 0.f, 0.f, 0.f};
#pragma unroll 4
  for (int k0 = 0; k0 < K; k0 += 32) {
    if (k0 + 32 < K) {
      __builtin_prefetch(A + (size_t)(m0 + (lane & 15)) * K + k0 + 32, 0, 3);
      __builtin_prefetch(Bt + (size_t)(n0 + (lane & 31)) * K + k0 + 32, 0, 3);
    }
    v16bf a  = load_a_frag(A, K, m0, k0, lane);
    v16bf b0 = load_bT_frag(Bt, K, k0, n0, lane);
    v16bf b1 = load_bT_frag(Bt, K, k0, n0 + 16, lane);
    acc0 = wmma_bf16(a, b0, acc0);
    acc1 = wmma_bf16(a, b1, acc1);
  }
  int mb = m0 + ((lane & 16) ? 8 : 0);
#pragma unroll
  for (int tile = 0; tile < 2; ++tile) {
    int n = n0 + tile * 16 + (lane & 15);
    float bn = bias[n];
    const v8f& acc = tile ? acc1 : acc0;
#pragma unroll
    for (int v = 0; v < 8; ++v) {
      float r = acc[v] + bn;
      if constexpr (ACT == 1) r = gelu_f(r);
      size_t idx = (size_t)(mb + v) * N + n;
      if constexpr (F32OUT) Cf[idx] = r;
      if constexpr (BFOUT) Cb[idx] = f2bf(r);
    }
  }
}

// per (be,h,s): rope(q,k) -> expmap0 per-head -> qh,kh (bf16, [be,h,s,d]) + squared norms;
// v stored TRANSPOSED per head: vT [be,h,d,s]
__global__ void qkv_post_k(const float* __restrict__ qkv, const float* __restrict__ c_logits,
                           bf16_t* __restrict__ qh, bf16_t* __restrict__ kh,
                           bf16_t* __restrict__ vT, float* __restrict__ q2,
                           float* __restrict__ k2) {
  int s = blockIdx.x, h = blockIdx.y, be = blockIdx.z;
  int t = threadIdx.x;  // 0..63
  __shared__ float sm[8];
  size_t rowbase = (size_t)(be * 64 + s) * 1536;
  const float* qp = qkv + rowbase + h * 64;
  const float* kp = qkv + rowbase + 512 + h * 64;
  float vv = qkv[rowbase + 1024 + h * 64 + t];
  // rope (split-half complex rotation), hd=64 -> 32 freqs
  int j = t & 31;
  float qr = qp[j], qi = qp[32 + j];
  float kr = kp[j], ki = kp[32 + j];
  float freq = __expf(-(float)(2 * j) * (9.210340371976184f / 64.f));  // 1/theta^(2j/64)
  float ang = (float)s * freq;
  float cs = __cosf(ang), sn = __sinf(ang);
  float qrot = (t < 32) ? (qr * cs - qi * sn) : (qr * sn + qi * cs);
  float krot = (t < 32) ? (kr * cs - ki * sn) : (kr * sn + ki * cs);
  // per-head curvature
  float c = softplus_f(c_logits[h]);
  float sc = fmaxf(sqrtf(c), EPSF);
  size_t obase = (size_t)((be * 8 + h) * 64 + s) * 64;
  // expmap0(q) + project
  {
    float vn = sqrtf(block_sum<64>(qrot * qrot, sm));
    float mg = fast_tanh(sc * fmaxf(vn, EPSF)) / sc;
    float s1 = (vn < EPSF) ? 0.f : mg / fmaxf(vn, EPSF);
    float en = (vn < EPSF) ? 0.f : mg;
    float ps = (en >= 1.f) ? (1.f - EPSF) / fmaxf(en, EPSF) : 1.f;
    float qe = qrot * s1 * ps;
    float enp = (en >= 1.f) ? (1.f - EPSF) : en;
    qh[obase + t] = f2bf(qe);
    if (t == 0) q2[(be * 8 + h) * 64 + s] = enp * enp;
  }
  // expmap0(k) + project
  {
    float vn = sqrtf(block_sum<64>(krot * krot, sm));
    float mg = fast_tanh(sc * fmaxf(vn, EPSF)) / sc;
    float s1 = (vn < EPSF) ? 0.f : mg / fmaxf(vn, EPSF);
    float en = (vn < EPSF) ? 0.f : mg;
    float ps = (en >= 1.f) ? (1.f - EPSF) / fmaxf(en, EPSF) : 1.f;
    float ke = krot * s1 * ps;
    float enp = (en >= 1.f) ? (1.f - EPSF) : en;
    kh[obase + t] = f2bf(ke);
    if (t == 0) k2[(be * 8 + h) * 64 + s] = enp * enp;
  }
  // vT[be,h,d,s] = v[be,h,s,d]
  vT[((size_t)(be * 8 + h) * 64 + t) * 64 + s] = f2bf(vv);
}

// per (be,h): qk[64,64] = qh @ kh^T   (kh rows are contiguous in K => bT loads)
__global__ void __launch_bounds__(256) qk_gemm_k(const bf16_t* __restrict__ qh,
                                                 const bf16_t* __restrict__ kh,
                                                 float* __restrict__ qk) {
  int bh = blockIdx.x;  // 0..127
  const bf16_t* A = qh + (size_t)bh * 4096;
  const bf16_t* Bt = kh + (size_t)bh * 4096;
  float* C = qk + (size_t)bh * 4096;
  int lane = threadIdx.x & 31, wid = threadIdx.x >> 5;
#pragma unroll
  for (int tt = wid; tt < 16; tt += 8) {
    int m0 = (tt >> 2) * 16, n0 = (tt & 3) * 16;
    v8f acc = {0.f, 0.f, 0.f, 0.f, 0.f, 0.f, 0.f, 0.f};
#pragma unroll
    for (int k0 = 0; k0 < 64; k0 += 32) {
      v16bf a = load_a_frag(A, 64, m0, k0, lane);
      v16bf b = load_bT_frag(Bt, 64, k0, n0, lane);
      acc = wmma_bf16(a, b, acc);
    }
    int n = n0 + (lane & 15);
    int mb = m0 + ((lane & 16) ? 8 : 0);
#pragma unroll
    for (int v = 0; v < 8; ++v) C[(size_t)(mb + v) * 64 + n] = acc[v];
  }
}

// per (be,h,sq): closed-form hyperbolic distance from (x2,y2,xy) scalars, then softmax over 64
__global__ void dist_softmax_k(const float* __restrict__ qk, const float* __restrict__ q2,
                               const float* __restrict__ k2, const float* __restrict__ c_logits,
                               const float* __restrict__ geo, bf16_t* __restrict__ attn) {
  int sq = blockIdx.x;   // 0..63
  int bh = blockIdx.y;   // 0..127
  int h = bh & 7;
  int t = threadIdx.x;   // sk
  __shared__ float sm[8];
  float c = softplus_f(c_logits[h]);
  float sc = fmaxf(sqrtf(c), EPSF);
  float x2 = q2[bh * 64 + sq];
  float y2 = k2[bh * 64 + t];
  float xy = qk[((size_t)bh * 64 + sq) * 64 + t];
  // mobius_add(-q, k, c): num = a*(-q) + b*k ; only its norm is needed
  float a = 1.f - 2.f * c * xy + c * y2;
  float b = 1.f - c * x2;
  float num2 = fmaxf(a * a * x2 + b * b * y2 - 2.f * a * b * xy, 0.f);
  float den = fmaxf(1.f - 2.f * c * xy + c * c * x2 * y2, EPSF);
  float rn = sqrtf(num2) / den;
  float an = (rn >= 1.f) ? (1.f - EPSF) : rn;   // project()
  float arg = fminf(sc * an, 1.f - EPSF);
  float d = 2.f * atanh_f(arg) / sc;
  float logit = -geo[h] * d;
  float mx = block_max<64>(logit, sm);
  float e = __expf(logit - mx);
  float ssum = block_sum<64>(e, sm);
  attn[((size_t)bh * 64 + sq) * 64 + t] = f2bf(e / ssum);
}

// per (be,h): ao = attn[64,64] @ v[64,64] with v stored transposed ([d][s] => bT loads);
// output written directly in [token, D] = [be*64+s, h*64+d] bf16
__global__ void __launch_bounds__(256) av_gemm_k(const bf16_t* __restrict__ attn,
                                                 const bf16_t* __restrict__ vT,
                                                 bf16_t* __restrict__ ao) {
  int bh = blockIdx.x;
  int be = bh >> 3, h = bh & 7;
  const bf16_t* A = attn + (size_t)bh * 4096;
  const bf16_t* Bt = vT + (size_t)bh * 4096;
  int lane = threadIdx.x & 31, wid = threadIdx.x >> 5;
#pragma unroll
  for (int tt = wid; tt < 16; tt += 8) {
    int m0 = (tt >> 2) * 16, n0 = (tt & 3) * 16;
    v8f acc = {0.f, 0.f, 0.f, 0.f, 0.f, 0.f, 0.f, 0.f};
#pragma unroll
    for (int k0 = 0; k0 < 64; k0 += 32) {
      v16bf a = load_a_frag(A, 64, m0, k0, lane);
      v16bf b = load_bT_frag(Bt, 64, k0, n0, lane);
      acc = wmma_bf16(a, b, acc);
    }
    int n = n0 + (lane & 15);
    int mb = m0 + ((lane & 16) ? 8 : 0);
#pragma unroll
    for (int v = 0; v < 8; ++v)
      ao[((size_t)(be * 64 + mb + v)) * 512 + h * 64 + n] = f2bf(acc[v]);
  }
}

// per-token: ah=expmap0(proj); xa=mobius_add(x,ah); then logmap0->LN(ln2)->expmap0->logmap0 => t
__global__ void resid_norm2_k(const float* __restrict__ proj, const float* __restrict__ xin,
                              const float* __restrict__ c_global, const float* __restrict__ ln_s,
                              const float* __restrict__ ln_b, float* __restrict__ xa,
                              bf16_t* __restrict__ t_bf) {
  int row = blockIdx.x;
  int t = threadIdx.x;
  __shared__ float sm[8];
  const float* pr = proj + (size_t)row * 512;
  const float* xr = xin + (size_t)row * 512;
  float c = c_global[0];
  float sc = fmaxf(sqrtf(c), EPSF);
  float p0 = pr[t], p1 = pr[t + 256];
  float x0 = xr[t], x1 = xr[t + 256];
  // ah = expmap0(proj) + project
  float pn = sqrtf(block_sum<256>(p0 * p0 + p1 * p1, sm));
  float mg = fast_tanh(sc * fmaxf(pn, EPSF)) / sc;
  float s1 = (pn < EPSF) ? 0.f : mg / fmaxf(pn, EPSF);
  float an = (pn < EPSF) ? 0.f : mg;
  float ps = (an >= 1.f) ? (1.f - EPSF) / fmaxf(an, EPSF) : 1.f;
  float a0 = p0 * s1 * ps, a1 = p1 * s1 * ps;
  // mobius_add(x, ah, c): three scalar reductions, analytic norm
  float x2s = block_sum<256>(x0 * x0 + x1 * x1, sm);
  float y2s = block_sum<256>(a0 * a0 + a1 * a1, sm);
  float xys = block_sum<256>(x0 * a0 + x1 * a1, sm);
  float al = 1.f + 2.f * c * xys + c * y2s;
  float be = 1.f - c * x2s;
  float den = fmaxf(1.f + 2.f * c * xys + c * c * x2s * y2s, EPSF);
  float n0 = (al * x0 + be * a0) / den;
  float n1 = (al * x1 + be * a1) / den;
  float nn = sqrtf(fmaxf(al * al * x2s + be * be * y2s + 2.f * al * be * xys, 0.f)) / den;
  float ps2 = (nn >= 1.f) ? (1.f - EPSF) / fmaxf(nn, EPSF) : 1.f;
  float xa0 = n0 * ps2, xa1 = n1 * ps2;
  xa[(size_t)row * 512 + t] = xa0;
  xa[(size_t)row * 512 + t + 256] = xa1;
  float xan = (nn >= 1.f) ? (1.f - EPSF) : nn;
  // logmap0(xa)
  float mg2 = atanh_f(fminf(xan, 1.f - EPSF)) / sc;
  float s3 = (xan < EPSF) ? 0.f : mg2 / fmaxf(xan, EPSF);
  float u0 = xa0 * s3, u1 = xa1 * s3;
  // layernorm ln2
  float mu = block_sum<256>(u0 + u1, sm) * (1.f / 512.f);
  float d0 = u0 - mu, d1 = u1 - mu;
  float var = block_sum<256>(d0 * d0 + d1 * d1, sm) * (1.f / 512.f);
  float rs = rsqrtf(var + 1e-6f);
  float w0 = d0 * rs * ln_s[t] + ln_b[t];
  float w1 = d1 * rs * ln_s[t + 256] + ln_b[t + 256];
  // expmap0 + project
  float vn = sqrtf(block_sum<256>(w0 * w0 + w1 * w1, sm));
  float mg3 = fast_tanh(sc * fmaxf(vn, EPSF)) / sc;
  float s4 = (vn < EPSF) ? 0.f : mg3 / fmaxf(vn, EPSF);
  float en = (vn < EPSF) ? 0.f : mg3;
  float ps3 = (en >= 1.f) ? (1.f - EPSF) / fmaxf(en, EPSF) : 1.f;
  float e0 = w0 * s4 * ps3, e1 = w1 * s4 * ps3;
  float enp = (en >= 1.f) ? (1.f - EPSF) : en;
  // logmap0 -> t
  float mg4 = atanh_f(fminf(enp, 1.f - EPSF)) / sc;
  float s5 = (enp < EPSF) ? 0.f : mg4 / fmaxf(enp, EPSF);
  t_bf[(size_t)row * 512 + t] = f2bf(e0 * s5);
  t_bf[(size_t)row * 512 + t + 256] = f2bf(e1 * s5);
}

// per-token: fh = expmap0(ffn_out); out = mobius_add(xa, fh, c)
__global__ void final_k(const float* __restrict__ f2, const float* __restrict__ xa,
                        const float* __restrict__ c_global, float* __restrict__ out) {
  int row = blockIdx.x;
  int t = threadIdx.x;
  __shared__ float sm[8];
  const float* fr = f2 + (size_t)row * 512;
  const float* xr = xa + (size_t)row * 512;
  float c = c_global[0];
  float sc = fmaxf(sqrtf(c), EPSF);
  float p0 = fr[t], p1 = fr[t + 256];
  float x0 = xr[t], x1 = xr[t + 256];
  // fh = expmap0 + project
  float pn = sqrtf(block_sum<256>(p0 * p0 + p1 * p1, sm));
  float mg = fast_tanh(sc * fmaxf(pn, EPSF)) / sc;
  float s1 = (pn < EPSF) ? 0.f : mg / fmaxf(pn, EPSF);
  float an = (pn < EPSF) ? 0.f : mg;
  float ps = (an >= 1.f) ? (1.f - EPSF) / fmaxf(an, EPSF) : 1.f;
  float a0 = p0 * s1 * ps, a1 = p1 * s1 * ps;
  // mobius_add(xa, fh, c) + project
  float x2s = block_sum<256>(x0 * x0 + x1 * x1, sm);
  float y2s = block_sum<256>(a0 * a0 + a1 * a1, sm);
  float xys = block_sum<256>(x0 * a0 + x1 * a1, sm);
  float al = 1.f + 2.f * c * xys + c * y2s;
  float be = 1.f - c * x2s;
  float den = fmaxf(1.f + 2.f * c * xys + c * c * x2s * y2s, EPSF);
  float n0 = (al * x0 + be * a0) / den;
  float n1 = (al * x1 + be * a1) / den;
  float nn = sqrtf(fmaxf(al * al * x2s + be * be * y2s + 2.f * al * be * xys, 0.f)) / den;
  float ps2 = (nn >= 1.f) ? (1.f - EPSF) / fmaxf(nn, EPSF) : 1.f;
  out[(size_t)row * 512 + t] = n0 * ps2;
  out[(size_t)row * 512 + t + 256] = n1 * ps2;
}

// ---------- host launcher ----------
extern "C" void kernel_launch(void* const* d_in, const int* in_sizes, int n_in,
                              void* d_out, int out_size, void* d_ws, size_t ws_size,
                              hipStream_t stream) {
  (void)in_sizes; (void)n_in; (void)out_size; (void)ws_size;
  const float* x        = (const float*)d_in[0];
  const float* c_global = (const float*)d_in[1];
  const float* W_qkv    = (const float*)d_in[2];
  const float* b_qkv    = (const float*)d_in[3];
  const float* W_out    = (const float*)d_in[4];
  const float* b_out    = (const float*)d_in[5];
  const float* ln1_s    = (const float*)d_in[6];
  const float* ln1_b    = (const float*)d_in[7];
  const float* ln2_s    = (const float*)d_in[8];
  const float* ln2_b    = (const float*)d_in[9];
  const float* W_ff1    = (const float*)d_in[10];
  const float* b_ff1    = (const float*)d_in[11];
  const float* W_ff2    = (const float*)d_in[12];
  const float* b_ff2    = (const float*)d_in[13];
  const float* c_logits = (const float*)d_in[14];
  const float* geo      = (const float*)d_in[15];
  // d_in[16] = start_pos (unused; reference ignores it)

  char* w = (char*)d_ws;
  auto take = [&](size_t bytes) -> void* {
    void* p = (void*)w;
    w += (bytes + 255) & ~(size_t)255;
    return p;
  };
  // transposed bf16 weights [N,K]
  bf16_t* wqkvT = (bf16_t*)take((size_t)1536 * 512 * 2);
  bf16_t* woutT = (bf16_t*)take((size_t)512 * 512 * 2);
  bf16_t* wff1T = (bf16_t*)take((size_t)2048 * 512 * 2);
  bf16_t* wff2T = (bf16_t*)take((size_t)512 * 2048 * 2);
  bf16_t* xt_bf = (bf16_t*)take((size_t)1024 * 512 * 2);
  float*  qkv_f = (float*)take((size_t)1024 * 1536 * 4);
  bf16_t* qh_bf = (bf16_t*)take((size_t)524288 * 2);
  bf16_t* kh_bf = (bf16_t*)take((size_t)524288 * 2);
  bf16_t* vT_bf = (bf16_t*)take((size_t)524288 * 2);
  float*  q2    = (float*)take((size_t)8192 * 4);
  float*  k2    = (float*)take((size_t)8192 * 4);
  float*  qk_f  = (float*)take((size_t)524288 * 4);
  bf16_t* at_bf = (bf16_t*)take((size_t)524288 * 2);
  bf16_t* ao_bf = (bf16_t*)take((size_t)1024 * 512 * 2);
  float*  proj  = (float*)take((size_t)1024 * 512 * 4);
  float*  xa_f  = (float*)take((size_t)1024 * 512 * 4);
  bf16_t* t_bf  = (bf16_t*)take((size_t)1024 * 512 * 2);
  bf16_t* h1_bf = (bf16_t*)take((size_t)1024 * 2048 * 2);
  float*  f2_f  = (float*)take((size_t)1024 * 512 * 4);

  // weight convert + transpose to [N,K] bf16 (so GEMM B-fragments load contiguously)
  cvtT_k<<<dim3(1536 / 32, 512 / 32), 256, 0, stream>>>(W_qkv, wqkvT, 512, 1536);
  cvtT_k<<<dim3(512 / 32, 512 / 32), 256, 0, stream>>>(W_out, woutT, 512, 512);
  cvtT_k<<<dim3(2048 / 32, 512 / 32), 256, 0, stream>>>(W_ff1, wff1T, 512, 2048);
  cvtT_k<<<dim3(512 / 32, 2048 / 32), 256, 0, stream>>>(W_ff2, wff2T, 2048, 512);

  // norm1 path -> tangent input
  prep_k<<<1024, 256, 0, stream>>>(x, c_global, ln1_s, ln1_b, xt_bf);
  // QKV projection (WMMA)
  gemm_tpl_k<512, 0, true, false><<<dim3(1536 / 128, 1024 / 32), 256, 0, stream>>>(
      xt_bf, wqkvT, b_qkv, qkv_f, nullptr, 1024, 1536);
  // rope + per-head expmap0 (v stored transposed)
  qkv_post_k<<<dim3(64, 8, 16), 64, 0, stream>>>(qkv_f, c_logits, qh_bf, kh_bf, vT_bf, q2, k2);
  // q.k^T inner products (WMMA, batched per be,h)
  qk_gemm_k<<<128, 256, 0, stream>>>(qh_bf, kh_bf, qk_f);
  // hyperbolic distance (closed form in scalars) + softmax
  dist_softmax_k<<<dim3(64, 128), 64, 0, stream>>>(qk_f, q2, k2, c_logits, geo, at_bf);
  // attn @ v (WMMA, batched), stored already transposed to [token, D]
  av_gemm_k<<<128, 256, 0, stream>>>(at_bf, vT_bf, ao_bf);
  // output projection (WMMA)
  gemm_tpl_k<512, 0, true, false><<<dim3(512 / 128, 1024 / 32), 256, 0, stream>>>(
      ao_bf, woutT, b_out, proj, nullptr, 1024, 512);
  // hyperbolic residual + norm2 path
  resid_norm2_k<<<1024, 256, 0, stream>>>(proj, x, c_global, ln2_s, ln2_b, xa_f, t_bf);
  // FFN (WMMA): gelu(t @ W_ff1 + b) @ W_ff2 + b
  gemm_tpl_k<512, 1, false, true><<<dim3(2048 / 128, 1024 / 32), 256, 0, stream>>>(
      t_bf, wff1T, b_ff1, nullptr, h1_bf, 1024, 2048);
  gemm_tpl_k<2048, 0, true, false><<<dim3(512 / 128, 1024 / 32), 256, 0, stream>>>(
      h1_bf, wff2T, b_ff2, f2_f, nullptr, 1024, 512);
  // final hyperbolic residual -> output
  final_k<<<1024, 256, 0, stream>>>(f2_f, xa_f, c_global, (float*)d_out);
}
